// SimplifiedTopologyExtractor_56650618635158
// MI455X (gfx1250) — matrix operand order repaired and checked by reading, not compiled, not run on hardware
//
#include <hip/hip_runtime.h>
#include <hip/hip_bf16.h>

// Fused kernel for SimplifiedTopologyExtractor on gfx1250 (MI455X).
//
// The reference's topology branch (W_proj projection, cdist, top_k, gather)
// is provably dead: pooled == embeddings. The live computation is
//   h   = embeddings[..., :256] @ W_enc + b_enc    (M=16384, K=256, N=512)
//   out = relu(LN(h) * gamma + beta)
//
// Roofline: ~51 MB of mandatory HBM traffic @ 23.3 TB/s ~= 2.2 us, vs ~6.6 us
// on the fp32 (16x16x4) WMMA pipe. So we run the GEMM in bf16 on
// V_WMMA_F32_16X16X32_BF16 (fp32 accumulate): W_enc is converted ONCE per call
// into d_ws as a [512][256] bf16 tensor in B-fragment order (two contiguous
// b128 loads per 16x32 fragment), A is converted to bf16 while staging into
// LDS. Bias + LayerNorm + ReLU stay fp32 and are fused so the 33.5 MB output
// is written exactly once.

typedef __attribute__((ext_vector_type(4)))  float  v4f;
typedef __attribute__((ext_vector_type(8)))  float  v8f;
typedef __attribute__((ext_vector_type(8)))  __bf16 v8bf;
typedef __attribute__((ext_vector_type(16))) __bf16 v16bf;

#define M_TOTAL        16384   // B*S = 8*2048
#define D_OUT          512
#define KDIM           256     // Dh
#define ROWS_PER_BLOCK 32
#define A_STRIDE       264     // bf16 elems: 256 + 8 pad (528B rows, 16B aligned)
#define H_STRIDE       516     // f32 elems: 512 + 4 pad
#define LN_EPS         1e-5f

static __device__ __forceinline__ unsigned pack_bf16(float a, float b) {
    union { __bf16 h[2]; unsigned u; } p;
    p.h[0] = (__bf16)a;   // RNE fptrunc
    p.h[1] = (__bf16)b;
    return p.u;
}

// ---------------------------------------------------------------------------
// One-time (per call) W_enc [K=256][N=512] f32 -> Bpack [N=512][K=256] bf16.
// Bpack row n holds column n of W_enc, so a 32-K B fragment is 64 contiguous
// bytes: dword j = K pair 2j,2j+1; lane half h loads K 8h+0..7 and 16+8h+0..7.
// Reads are coalesced across threads (fixed k, consecutive n). 0.75 MB total.
// ---------------------------------------------------------------------------
__global__ __launch_bounds__(256)
void pack_wenc_bf16_kernel(const float* __restrict__ Wenc,
                           __bf16* __restrict__ Bpack)
{
    const int n = blockIdx.x * 256 + threadIdx.x;   // 0..511
    unsigned* dst = (unsigned*)(Bpack + (size_t)n * KDIM);
    #pragma unroll 8
    for (int k = 0; k < KDIM; k += 2) {
        float a = Wenc[(size_t)k * D_OUT + n];
        float b = Wenc[(size_t)(k + 1) * D_OUT + n];
        dst[k >> 1] = pack_bf16(a, b);
    }
}

// ---------------------------------------------------------------------------
// Main fused kernel: 512 blocks x 256 threads (8 wave32 waves).
// Block owns 32 rows x all 512 cols (so LayerNorm fuses); wave owns 64 cols.
// ---------------------------------------------------------------------------
__global__ __launch_bounds__(256)
void topo_enc_ln_relu_kernel(const float* __restrict__ emb,
                             const __bf16* __restrict__ Bpack,
                             const float* __restrict__ benc,
                             const float* __restrict__ gamma,
                             const float* __restrict__ beta,
                             float* __restrict__ out)
{
    __shared__ __bf16 Ash[ROWS_PER_BLOCK * A_STRIDE];   // 16.5 KB
    __shared__ float  Hsh[ROWS_PER_BLOCK * H_STRIDE];   // 66.0 KB

    const int tid  = threadIdx.x;
    const int wave = tid >> 5;          // 8 wave32 waves per block
    const int lane = tid & 31;
    const int half = lane >> 4;         // 0: lanes 0-15, 1: lanes 16-31
    const int nl   = lane & 15;
    const int rowBase = blockIdx.x * ROWS_PER_BLOCK;

    // ---- Stage A tile: 16 rows x 256 cols of embeddings, f32 -> bf16 ------
    // 1024 float4 chunks over 256 threads: b128 global load -> cvt -> b64 ds.
    #pragma unroll
    for (int it = 0; it < 4; ++it) {
        int idx = it * 256 + tid;
        int r   = idx >> 6;             // 64 float4 per row
        int c4  = idx & 63;
        v4f v = *(const v4f*)(emb + (size_t)(rowBase + r) * D_OUT + c4 * 4);
        unsigned* dst = (unsigned*)(Ash + r * A_STRIDE + c4 * 4);
        dst[0] = pack_bf16(v.x, v.y);
        dst[1] = pack_bf16(v.z, v.w);
    }
    __syncthreads();

    // ---- bf16 GEMM: wave owns N in [wave*64, wave*64+64), both M halves ---
    const int n0w = wave * 64;
    v8f acc[2][4] = {};

    // Per-lane B base: column n = n0w + t*16 + nl; half adds K offset 8 (K 0-7,16-23 / 8-15,24-31).
    const __bf16* bbase[4];
    #pragma unroll
    for (int t = 0; t < 4; ++t)
        bbase[t] = Bpack + (size_t)(n0w + t * 16 + nl) * KDIM + 8 * half;

    #pragma unroll
    for (int k0 = 0; k0 < KDIM; k0 += 32) {
        if (k0 + 32 < KDIM)   // uniform branch: EXEC untouched (WMMA-safe)
            __builtin_prefetch(bbase[0] + k0 + 32, 0, 3);

        // A fragments (16-bit 16x32 layout): lane row = m*16 + nl,
        // lanes 0-15: K = k0+0..7 then k0+16..23; lanes 16-31: +8.
        v16bf afrag[2];
        #pragma unroll
        for (int m = 0; m < 2; ++m) {
            const __bf16* ap = Ash + (m * 16 + nl) * A_STRIDE + 8 * half + k0;
            v8bf lo = *(const v8bf*)(ap);
            v8bf hi = *(const v8bf*)(ap + 16);
            afrag[m] = __builtin_shufflevector(lo, hi,
                0, 1, 2, 3, 4, 5, 6, 7, 8, 9, 10, 11, 12, 13, 14, 15);
        }

        #pragma unroll
        for (int t = 0; t < 4; ++t) {
            // B fragment: two 8-bf16 runs of Bpack row n (K +0..7, +16..23) -> two b128 loads.
            const __bf16* bp = bbase[t] + k0;
            v8bf lo = *(const v8bf*)(bp);
            v8bf hi = *(const v8bf*)(bp + 16);
            v16bf bfrag = __builtin_shufflevector(lo, hi,
                0, 1, 2, 3, 4, 5, 6, 7, 8, 9, 10, 11, 12, 13, 14, 15);
            #pragma unroll
            for (int m = 0; m < 2; ++m)
                acc[m][t] = __builtin_amdgcn_wmma_f32_16x16x32_bf16(
                    false, afrag[m], false, bfrag,
                    (short)0, acc[m][t], false, false);
        }
    }

    // ---- Epilogue part 1: bias add, scatter C tiles into LDS --------------
    // C layout: VGPR j -> row M = j + 8*half, col N = n0 + nl.
    #pragma unroll
    for (int t = 0; t < 4; ++t) {
        float bv = benc[n0w + t * 16 + nl];
        #pragma unroll
        for (int m = 0; m < 2; ++m) {
            #pragma unroll
            for (int j = 0; j < 8; ++j) {
                int row = m * 16 + j + 8 * half;
                Hsh[row * H_STRIDE + n0w + t * 16 + nl] = acc[m][t][j] + bv;
            }
        }
    }
    __syncthreads();

    // ---- Epilogue part 2: LayerNorm + ReLU. 8 threads per row, 32 rows ----
    const int row = tid >> 3;
    const int tr  = tid & 7;
    float sum = 0.f, ss = 0.f;
    #pragma unroll
    for (int i = 0; i < 16; ++i) {
        int col = (tr + 8 * i) * 4;
        v4f v = *(const v4f*)(Hsh + row * H_STRIDE + col);
        sum += v.x + v.y + v.z + v.w;
        ss  += v.x * v.x + v.y * v.y + v.z * v.z + v.w * v.w;
    }
    // width-8 xor-reduction within the wave32 (lanes 8k..8k+7 share a row)
    #pragma unroll
    for (int m = 1; m < 8; m <<= 1) {
        sum += __shfl_xor(sum, m, 8);
        ss  += __shfl_xor(ss,  m, 8);
    }
    const float mean = sum * (1.0f / (float)D_OUT);
    const float var  = ss  * (1.0f / (float)D_OUT) - mean * mean;
    const float rstd = rsqrtf(var + LN_EPS);

    float* orow = out + (size_t)(rowBase + row) * D_OUT;
    #pragma unroll
    for (int i = 0; i < 16; ++i) {
        int col = (tr + 8 * i) * 4;
        v4f h = *(const v4f*)(Hsh + row * H_STRIDE + col);
        v4f g = *(const v4f*)(gamma + col);
        v4f b = *(const v4f*)(beta  + col);
        v4f o;
        o.x = fmaxf((h.x - mean) * rstd * g.x + b.x, 0.f);
        o.y = fmaxf((h.y - mean) * rstd * g.y + b.y, 0.f);
        o.z = fmaxf((h.z - mean) * rstd * g.z + b.z, 0.f);
        o.w = fmaxf((h.w - mean) * rstd * g.w + b.w, 0.f);
        *(v4f*)(orow + col) = o;
    }
}

extern "C" void kernel_launch(void* const* d_in, const int* in_sizes, int n_in,
                              void* d_out, int out_size, void* d_ws, size_t ws_size,
                              hipStream_t stream) {
    (void)in_sizes; (void)n_in; (void)ws_size; (void)out_size;
    // setup_inputs order: embeddings, W_proj, b_proj, W_enc, b_enc, ln_gamma, ln_beta
    // W_proj / b_proj are dead (output provably independent of them).
    const float* emb   = (const float*)d_in[0];
    const float* Wenc  = (const float*)d_in[3];
    const float* benc  = (const float*)d_in[4];
    const float* gamma = (const float*)d_in[5];
    const float* beta  = (const float*)d_in[6];
    float* out = (float*)d_out;

    // Workspace: Bpack = 512*256 bf16 = 256 KB (deterministic: rebuilt every call).
    __bf16* Bpack = (__bf16*)d_ws;

    pack_wenc_bf16_kernel<<<dim3(2), dim3(256), 0, stream>>>(Wenc, Bpack);

    dim3 grid(M_TOTAL / ROWS_PER_BLOCK);   // 512 blocks
    dim3 block(256);                        // 8 wave32 waves
    topo_enc_ln_relu_kernel<<<grid, block, 0, stream>>>(emb, Bpack, benc, gamma, beta, out);
}
